// Attention_1_23055384445180
// MI455X (gfx1250) — matrix operand order repaired and checked by reading, not compiled, not run on hardware
//
#include <hip/hip_runtime.h>
#include <hip/hip_bf16.h>

typedef unsigned short u16;
typedef unsigned int   u32;
typedef __attribute__((ext_vector_type(16))) __bf16 v16bf;
typedef __attribute__((ext_vector_type(8)))  float  v8f;
typedef __attribute__((ext_vector_type(4)))  float  v4f;
typedef __attribute__((ext_vector_type(4)))  u32    v4u;
typedef __attribute__((ext_vector_type(4)))  int    v4i;

// Pointer types for the async global->LDS builtin (param 0 is
// "int __vector(4) __device__ *" per the round-2 diagnostic).
typedef __attribute__((address_space(1))) v4i* as1_v4i_ptr;
typedef __attribute__((address_space(3))) v4i* as3_v4i_ptr;

#define HIDDEN   1024
#define HEADS    16
#define HEAD_DIM 64
#define SEQ      2048
#define BATCH    2
#define BS       (BATCH * SEQ)   // 4096 rows

#if __has_builtin(__builtin_amdgcn_global_load_async_to_lds_b128)
#define HAVE_ASYNC_LDS 1
#else
#define HAVE_ASYNC_LDS 0
#endif

union FragU {
    v16bf bf;
    u16   u[16];
    v4u   q[2];
};

__device__ __forceinline__ u16 f32_to_bf16(float f) {
    u32 x = __float_as_uint(f);
    x += 0x7FFFu + ((x >> 16) & 1u);   // round-to-nearest-even
    return (u16)(x >> 16);
}

__device__ __forceinline__ v8f wmma_bf16(v16bf a, v16bf b, v8f c) {
    // v_wmma_f32_16x16x32_bf16: 8 args (neg_a, A, neg_b, B, c_mod, C, reuse_a, reuse_b)
    return __builtin_amdgcn_wmma_f32_16x16x32_bf16(false, a, false, b, (short)0, c, false, false);
}

__device__ __forceinline__ void wait_async_zero() {
#if HAVE_ASYNC_LDS
#if __has_builtin(__builtin_amdgcn_s_wait_asynccnt)
    __builtin_amdgcn_s_wait_asynccnt(0);
#else
    asm volatile("s_wait_asynccnt 0x0" ::: "memory");
#endif
#endif
}

// ---- fragment loaders (wave32 layouts per CDNA5 ISA 7.12.2) -----------------

// A 16x32 bf16: lane = M (lane&15); half-wave kbase 0/8; chunks [kb..kb+7], [kb+16..kb+23]
__device__ __forceinline__ v16bf load_frag_a(const u16* __restrict__ A, int lda, int m0, int k0) {
    int lane = threadIdx.x & 31;
    const u16* p = A + (size_t)(m0 + (lane & 15)) * lda + k0 + ((lane < 16) ? 0 : 8);
    FragU f;
    f.q[0] = *(const v4u*)(p);
    f.q[1] = *(const v4u*)(p + 16);
    return f.bf;
}

// B 32x16 bf16 where B[k,n] = Bm[n,k] (Bm row-major [N,K]): lane = N; kbase 0/8;
// per-lane chunks [kb..kb+7], [kb+16..kb+23] of Bm row n.
__device__ __forceinline__ v16bf load_frag_b(const u16* __restrict__ Bm, int ldb, int n0, int k0) {
    int lane = threadIdx.x & 31;
    const u16* p = Bm + (size_t)(n0 + (lane & 15)) * ldb + k0 + ((lane < 16) ? 0 : 8);
    FragU f;
    f.q[0] = *(const v4u*)(p);
    f.q[1] = *(const v4u*)(p + 16);
    return f.bf;
}

// A 16x32 from fp32 source (attn), converted to bf16 in registers (vector loads)
__device__ __forceinline__ v16bf load_frag_a_f32(const float* __restrict__ A, int lda, int m0, int k0) {
    int lane = threadIdx.x & 31;
    const float* p = A + (size_t)(m0 + (lane & 15)) * lda + k0 + ((lane < 16) ? 0 : 8);
    v4f c0 = *(const v4f*)(p);
    v4f c1 = *(const v4f*)(p + 4);
    v4f c2 = *(const v4f*)(p + 16);
    v4f c3 = *(const v4f*)(p + 20);
    FragU f;
#pragma unroll
    for (int e = 0; e < 4; ++e) {
        f.u[e]      = f32_to_bf16(c0[e]);
        f.u[4 + e]  = f32_to_bf16(c1[e]);
        f.u[8 + e]  = f32_to_bf16(c2[e]);
        f.u[12 + e] = f32_to_bf16(c3[e]);
    }
    return f.bf;
}

// B 32x16 from LDS tile (row-major [32][64]): B[k,n] = tile[k][n0+n].
// Lane addresses are consecutive u16 across lanes -> bank-conflict-free.
__device__ __forceinline__ v16bf load_frag_b_lds(const u16* tile, int n0) {
    int lane = threadIdx.x & 31;
    int n  = n0 + (lane & 15);
    int kb = (lane < 16) ? 0 : 8;
    FragU f;
#pragma unroll
    for (int e = 0; e < 16; ++e) f.u[e] = tile[(kb + e + (e >= 8 ? 8 : 0)) * 64 + n];
    return f.bf;
}

// ---- kernels ----------------------------------------------------------------

__global__ void cvt_f32_to_bf16(const float* __restrict__ src, u16* __restrict__ dst, int n) {
    int i = blockIdx.x * blockDim.x + threadIdx.x;
    int stride = gridDim.x * blockDim.x;
    for (; i < n; i += stride) dst[i] = f32_to_bf16(src[i]);
}

// Y = X @ W^T.  X bf16 [M,Kd] row-major, W bf16 [N,Kd] row-major.
// mode 0: write bf16 Y.  mode 1: write fp32 sigmoid(Y + bias) (gate).
// Software-pipelined: next k-tile fragments are loaded before the current WMMA.
__global__ void proj_gemm_kernel(const u16* __restrict__ X, const u16* __restrict__ W,
                                 u16* __restrict__ Ybf, float* __restrict__ Ygate,
                                 const float* __restrict__ bias,
                                 int M, int N, int Kd, int mode) {
    int wave = threadIdx.x >> 5;
    int lane = threadIdx.x & 31;
    int m0 = blockIdx.x * 64 + (wave & 3) * 16;
    int n0 = blockIdx.y * 32 + (wave >> 2) * 16;

    v8f acc = {};
    v16bf a = load_frag_a(X, Kd, m0, 0);
    v16bf b = load_frag_b(W, Kd, n0, 0);
    for (int kk = 0; kk < Kd; kk += 32) {
        v16bf an = a, bn = b;
        if (kk + 64 < Kd) {
            __builtin_prefetch(X + (size_t)(m0 + (lane & 15)) * Kd + kk + 64, 0, 3);
            __builtin_prefetch(W + (size_t)(n0 + (lane & 15)) * Kd + kk + 64, 0, 3);
        }
        if (kk + 32 < Kd) {
            an = load_frag_a(X, Kd, m0, kk + 32);
            bn = load_frag_b(W, Kd, n0, kk + 32);
        }
        acc = wmma_bf16(a, b, acc);
        a = an; b = bn;
    }

    int n  = n0 + (lane & 15);
    int mb = m0 + ((lane < 16) ? 0 : 8);
    if (mode == 0) {
#pragma unroll
        for (int r = 0; r < 8; ++r)
            Ybf[(size_t)(mb + r) * N + n] = f32_to_bf16(acc[r]);
    } else {
        float bb = bias[n];
#pragma unroll
        for (int r = 0; r < 8; ++r) {
            float y = acc[r] + bb;
            Ygate[(size_t)(mb + r) * N + n] = 1.0f / (1.0f + __expf(-y));
        }
    }
}

// scores[z][i,j] = (1/8) * sum_d K[b*S+i, h*64+d] * Q[b*S+j, h*64+d], fp32 into attn region
__global__ void scores_kernel(const u16* __restrict__ Kbf, const u16* __restrict__ Qbf,
                              float* __restrict__ attn) {
    int z = blockIdx.z;                 // b*HEADS + h
    int b = z >> 4, h = z & 15;
    const u16* Kbase = Kbf + (size_t)b * SEQ * HIDDEN + h * HEAD_DIM;
    const u16* Qbase = Qbf + (size_t)b * SEQ * HIDDEN + h * HEAD_DIM;
    float* attnZ = attn + (size_t)z * SEQ * SEQ;
    int wave = threadIdx.x >> 5, lane = threadIdx.x & 31;
    int m0 = blockIdx.x * 64 + (wave & 3) * 16;
    int n0 = blockIdx.y * 32 + (wave >> 2) * 16;

    // HEAD_DIM = 64 -> exactly two k-steps; issue all loads, then both WMMAs
    v16bf a0 = load_frag_a(Kbase, HIDDEN, m0, 0);
    v16bf q0 = load_frag_b(Qbase, HIDDEN, n0, 0);
    v16bf a1 = load_frag_a(Kbase, HIDDEN, m0, 32);
    v16bf q1 = load_frag_b(Qbase, HIDDEN, n0, 32);
    v8f acc = {};
    acc = wmma_bf16(a0, q0, acc);
    acc = wmma_bf16(a1, q1, acc);

    int n  = n0 + (lane & 15);
    int mb = m0 + ((lane < 16) ? 0 : 8);
    const float scale = 0.125f;         // 1/sqrt(64)
#pragma unroll
    for (int r = 0; r < 8; ++r)
        attnZ[(size_t)(mb + r) * SEQ + n] = acc[r] * scale;
}

// row-wise softmax over last axis (q positions), in place; one block per row of 2048
__global__ void softmax_rows_kernel(float* __restrict__ attn) {
    __shared__ float red[256];
    float* row = attn + (size_t)blockIdx.x * SEQ;
    int t = threadIdx.x;
    float vals[8];
    float m = -3.4e38f;
#pragma unroll
    for (int i = 0; i < 8; ++i) { vals[i] = row[t + i * 256]; m = fmaxf(m, vals[i]); }
    red[t] = m; __syncthreads();
    for (int s = 128; s > 0; s >>= 1) { if (t < s) red[t] = fmaxf(red[t], red[t + s]); __syncthreads(); }
    m = red[0]; __syncthreads();
    float sum = 0.0f;
#pragma unroll
    for (int i = 0; i < 8; ++i) { vals[i] = __expf(vals[i] - m); sum += vals[i]; }
    red[t] = sum; __syncthreads();
    for (int s = 128; s > 0; s >>= 1) { if (t < s) red[t] += red[t + s]; __syncthreads(); }
    float inv = 1.0f / red[0];
#pragma unroll
    for (int i = 0; i < 8; ++i) row[t + i * 256] = vals[i] * inv;
}

// ctx[z][i,d] = sum_j attn[z][i,j] * V[b*S+j, h*64+d]  (V == K projection), fused gate.
// The shared 32x64 V-tile per k-step is staged in LDS: one coalesced b128 per thread
// (async global->LDS when available), shared by all 8 waves of the block.
__global__ void ctx_gate_kernel(const float* __restrict__ attn, const u16* __restrict__ Vbf,
                                const float* __restrict__ gate, float* __restrict__ out) {
    __shared__ __align__(16) u16 tile[32][64];   // 4 KB: V[jk..jk+31, hd..hd+63]

    int z = blockIdx.z;
    int b = z >> 4, h = z & 15;
    const float* attnZ = attn + (size_t)z * SEQ * SEQ;
    const u16* Vbase = Vbf + (size_t)b * SEQ * HIDDEN + h * HEAD_DIM;
    int wave = threadIdx.x >> 5, lane = threadIdx.x & 31;
    int m0 = blockIdx.x * 32 + (wave & 1) * 16;   // key-position tile
    int n0 = (wave >> 1) * 16;                    // head-dim tile (64 = 4 tiles)

    int t    = threadIdx.x;
    int srow = t >> 3;        // 0..31  (j within tile)
    int scol = (t & 7) * 8;   // 0..56  (d within head, 8 u16 = 16B per thread)

    v8f acc = {};
    for (int jk = 0; jk < SEQ; jk += 32) {
        __syncthreads();      // tile reads from previous iteration complete
        const u16* gsrc = Vbase + (size_t)(jk + srow) * HIDDEN + scol;
#if HAVE_ASYNC_LDS
        __builtin_amdgcn_global_load_async_to_lds_b128(
            (as1_v4i_ptr)gsrc,
            (as3_v4i_ptr)&tile[srow][scol],
            0, 0);
        wait_async_zero();
#else
        *(v4u*)&tile[srow][scol] = *(const v4u*)gsrc;
#endif
        __syncthreads();      // tile fully populated
        v16bf a = load_frag_a_f32(attnZ, SEQ, m0, jk);
        v16bf v = load_frag_b_lds(&tile[0][0], n0);
        acc = wmma_bf16(a, v, acc);
    }

    int n  = n0 + (lane & 15);
    int mb = m0 + ((lane < 16) ? 0 : 8);
#pragma unroll
    for (int r = 0; r < 8; ++r) {
        size_t row = (size_t)b * SEQ + mb + r;
        size_t col = (size_t)h * HEAD_DIM + n;
        out[row * HIDDEN + col] = gate[row * HIDDEN + col] * acc[r];
    }
}

// ---- host glue --------------------------------------------------------------

extern "C" void kernel_launch(void* const* d_in, const int* in_sizes, int n_in,
                              void* d_out, int out_size, void* d_ws, size_t ws_size,
                              hipStream_t stream) {
    const float* x  = (const float*)d_in[0];
    const float* Wq = (const float*)d_in[1];
    const float* Wk = (const float*)d_in[2];
    const float* Wg = (const float*)d_in[3];
    const float* bg = (const float*)d_in[4];

    char* ws = (char*)d_ws;
    u16*   xbf  = (u16*)(ws);                              //  8 MB  [0,8)
    u16*   wqbf = (u16*)(ws + (size_t)(8 << 20));          //  2 MB  [8,10)
    u16*   wkbf = (u16*)(ws + (size_t)(10 << 20));         //  2 MB  [10,12)
    u16*   wgbf = (u16*)(ws + (size_t)(12 << 20));         //  2 MB  [12,14)
    u16*   Qbf  = (u16*)(ws + (size_t)(14 << 20));         //  8 MB  [14,22)
    u16*   Kbf  = (u16*)(ws + (size_t)(22 << 20));         //  8 MB  [22,30)
    float* gate = (float*)(ws + (size_t)(30 << 20));       // 16 MB  [30,46)

    float* out  = (float*)d_out;                                   // [B,S,H] fp32
    float* attn = out + (size_t)BATCH * SEQ * HIDDEN;              // [B,h,S,S] fp32

    dim3 blk(256);

    cvt_f32_to_bf16<<<1024, blk, 0, stream>>>(x,  xbf,  BS * HIDDEN);
    cvt_f32_to_bf16<<<512,  blk, 0, stream>>>(Wq, wqbf, HIDDEN * HIDDEN);
    cvt_f32_to_bf16<<<512,  blk, 0, stream>>>(Wk, wkbf, HIDDEN * HIDDEN);
    cvt_f32_to_bf16<<<512,  blk, 0, stream>>>(Wg, wgbf, HIDDEN * HIDDEN);

    dim3 ggrid(BS / 64, HIDDEN / 32);
    proj_gemm_kernel<<<ggrid, blk, 0, stream>>>(xbf, wqbf, Qbf, nullptr, nullptr, BS, HIDDEN, HIDDEN, 0);
    proj_gemm_kernel<<<ggrid, blk, 0, stream>>>(xbf, wkbf, Kbf, nullptr, nullptr, BS, HIDDEN, HIDDEN, 0);
    proj_gemm_kernel<<<ggrid, blk, 0, stream>>>(xbf, wgbf, nullptr, gate, bg,     BS, HIDDEN, HIDDEN, 1);

    dim3 sgrid(SEQ / 64, SEQ / 32, BATCH * HEADS);
    scores_kernel<<<sgrid, blk, 0, stream>>>(Kbf, Qbf, attn);

    softmax_rows_kernel<<<BATCH * HEADS * SEQ, blk, 0, stream>>>(attn);

    dim3 cgrid(SEQ / 32, 1, BATCH * HEADS);
    ctx_gate_kernel<<<cgrid, blk, 0, stream>>>(attn, Kbf, gate, out);
}
